// SimpleAbsorbedAttention_60258391163523
// MI455X (gfx1250) — compile-verified, matrix operand order checked by reading, not run on hardware
//
#include <hip/hip_runtime.h>

// ---------------------------------------------------------------------------
// MLA absorbed-attention decode, MI455X (gfx1250, wave32, WMMA bf16).
//   BSZ=32, SEQ=4096, NH=128, HD=128, LORA=512, Q_LORA=1536
// Memory-bound (~0.44 GB unique, ~19us HBM floor); all GEMMs via
// v_wmma_f32_16x16x32_bf16 (f32->bf16 operands, f32 accum). Softmax done
// exactly in one pass using a 256KB LDS score tile (CDNA5 has 320KB/WGP).
// k_attn uses 8 waves/WGP (2 per SIMD32) to hide WMMA/VMEM latency.
// ---------------------------------------------------------------------------

#define BSZ   32
#define SEQ   4096
#define NH    128
#define HD    128
#define LORA  512
#define QLORA 1536
#define NHHD  (NH * HD)

typedef __bf16 bf16;
typedef __attribute__((ext_vector_type(16))) __bf16 v16bf;
typedef __attribute__((ext_vector_type(8)))  float  v8f;

// D = A(16x32 bf16) x B(32x16 bf16) + C(16x16 f32)
static __device__ __forceinline__ v8f wmma_bf16(v16bf a, v16bf b, v8f c) {
  return __builtin_amdgcn_wmma_f32_16x16x32_bf16(
      /*neg_a=*/false, a, /*neg_b=*/false, b,
      /*c_mod=*/(short)0, c, /*reuse_a=*/false, /*reuse_b=*/false);
}

// 16-bit A/B fragment layout (ISA 7.12.2): lane = M (or N) index in [0,16),
// half-wave selects K octets: lane holds K = h*8+[0..8) and K = 16+h*8+[0..8).
// `p` must already be offset by  row*ld + kbase + half*8.
static __device__ __forceinline__ v16bf frag_row_f32(const float* p) {
  v16bf f;
#pragma unroll
  for (int i = 0; i < 8; ++i) f[i] = (__bf16)p[i];
#pragma unroll
  for (int i = 0; i < 8; ++i) f[8 + i] = (__bf16)p[16 + i];
  return f;
}
static __device__ __forceinline__ v16bf frag_row_bf16(const bf16* p) {
  v16bf f;
#pragma unroll
  for (int i = 0; i < 8; ++i) f[i] = p[i];
#pragma unroll
  for (int i = 0; i < 8; ++i) f[8 + i] = p[16 + i];
  return f;
}
// Gather fragment: element K' -> base[(kofs + K')*ldk]. Across the 16 lanes of
// a half-wave the addresses are contiguous (64B segments), so these coalesce.
static __device__ __forceinline__ v16bf frag_strided_f32(const float* base,
                                                         int ldk, int kofs) {
  const float* p = base + (size_t)kofs * ldk;
  v16bf f;
#pragma unroll
  for (int i = 0; i < 8; ++i) f[i] = (__bf16)p[(size_t)i * ldk];
  const float* q = p + (size_t)16 * ldk;
#pragma unroll
  for (int i = 0; i < 8; ++i) f[8 + i] = (__bf16)q[(size_t)i * ldk];
  return f;
}

// ---------------------------------------------------------------------------
// Kernel 1: q = hs_q @ q_w^T + q_b   (M=32, N=16384, K=1536) -> bf16
// grid (256, 2), block 128 (4 waves; each wave owns one 16-wide N tile)
// ---------------------------------------------------------------------------
__global__ void k_qproj(const float* __restrict__ hs,
                        const float* __restrict__ qw,
                        const float* __restrict__ qb,
                        bf16* __restrict__ qbf) {
  const int lane = threadIdx.x & 31, wave = threadIdx.x >> 5;
  const int row = lane & 15, half = lane >> 4;
  const int n0 = (blockIdx.x * 4 + wave) * 16;  // 0..16368
  const int m0 = blockIdx.y * 16;               // 0 or 16

  const float* arow = hs + (size_t)(m0 + row) * QLORA + half * 8;
  const float* brow = qw + (size_t)(n0 + row) * QLORA + half * 8;  // K contiguous

  v8f acc = {};
#pragma unroll 4
  for (int kt = 0; kt < QLORA / 32; ++kt) {
    v16bf a = frag_row_f32(arow + kt * 32);
    v16bf b = frag_row_f32(brow + kt * 32);
    acc = wmma_bf16(a, b, acc);
  }
  const int n = n0 + (lane & 15);
  const int mb = m0 + (lane >> 4) * 8;
  const float bias = qb[n];
#pragma unroll
  for (int e = 0; e < 8; ++e)
    qbf[(size_t)(mb + e) * NHHD + n] = (bf16)(acc[e] + bias);
}

// ---------------------------------------------------------------------------
// Kernel 2: q_absorbed[b,h,l] = sum_d q[b,h,d] * w_kc[h,d,l]  -> bf16
// grid (8, 128, 2), block 128 (4 waves; each wave owns one 16-wide L tile)
// ---------------------------------------------------------------------------
__global__ void k_absorb(const bf16* __restrict__ qbf,
                         const float* __restrict__ wkc,
                         bf16* __restrict__ qa) {
  const int lane = threadIdx.x & 31, wave = threadIdx.x >> 5;
  const int row = lane & 15, half = lane >> 4;
  const int l0 = (blockIdx.x * 4 + wave) * 16;  // 0..511
  const int h = blockIdx.y;
  const int m0 = blockIdx.z * 16;

  const bf16*  arow  = qbf + (size_t)(m0 + row) * NHHD + (size_t)h * HD + half * 8;
  const float* bbase = wkc + (size_t)h * HD * LORA + (l0 + row);  // stride LORA over d

  v8f acc = {};
#pragma unroll
  for (int kt = 0; kt < HD / 32; ++kt) {   // 4 steps
    v16bf a = frag_row_bf16(arow + kt * 32);
    v16bf b = frag_strided_f32(bbase, LORA, kt * 32 + half * 8);
    acc = wmma_bf16(a, b, acc);
  }
  const int l = l0 + (lane & 15);
  const int mb = m0 + (lane >> 4) * 8;
#pragma unroll
  for (int e = 0; e < 8; ++e)
    qa[((size_t)(mb + e) * NH + h) * LORA + l] = (bf16)acc[e];
}

// ---------------------------------------------------------------------------
// Kernel 3: fused scores + exact softmax + context, one (batch, 16-head) tile
// per block. Dynamic LDS: 16 x 4096 f32 scores = 256 KB (fits CDNA5 320KB).
// grid (8, 32), block 256 (8 waves -> 2 waves per SIMD32 for latency hiding).
// ---------------------------------------------------------------------------
__global__ void k_attn(const bf16* __restrict__ qa,
                       const float* __restrict__ kv,
                       float* __restrict__ ctx) {
  extern __shared__ float sS[];  // [16][SEQ]
  const int lane = threadIdx.x & 31, wave = threadIdx.x >> 5;  // wave 0..7
  const int row = lane & 15, half = lane >> 4;
  const int h0 = blockIdx.x * 16;
  const int b  = blockIdx.y;
  const float* kvb = kv + (size_t)b * SEQ * LORA;

  // Preload A fragments: q_absorbed [16h x 512l] bf16 -> 16 frags in VGPRs.
  const bf16* qarow = qa + ((size_t)b * NH + h0 + row) * LORA + half * 8;
  v16bf aF[16];
#pragma unroll
  for (int kt = 0; kt < 16; ++kt) aF[kt] = frag_row_bf16(qarow + kt * 32);

  // ---- Phase 1: S[h,s] = sum_l QA[h,l]*KV[s,l]; 8 waves split seq tiles ----
  for (int st = wave; st < SEQ / 16; st += 8) {
    const float* brow = kvb + (size_t)(st * 16 + row) * LORA + half * 8;
    if (st + 8 < SEQ / 16)  // stream-ahead prefetch of next KV rows
      __builtin_prefetch(kvb + (size_t)((st + 8) * 16 + row) * LORA, 0, 1);
    v8f acc = {};
#pragma unroll
    for (int kt = 0; kt < 16; ++kt)
      acc = wmma_bf16(aF[kt], frag_row_f32(brow + kt * 32), acc);
    const int n  = st * 16 + (lane & 15);
    const int mb = (lane >> 4) * 8;
#pragma unroll
    for (int e = 0; e < 8; ++e) sS[(size_t)(mb + e) * SEQ + n] = acc[e];
  }
  __syncthreads();

  // ---- Phase 2: exact softmax per head row (wave32 shuffle reductions) ----
  for (int r = wave; r < 16; r += 8) {
    float* srow = sS + (size_t)r * SEQ;
    float mx = -3.0e38f;
    for (int s = lane; s < SEQ; s += 32) mx = fmaxf(mx, srow[s]);
#pragma unroll
    for (int o = 16; o > 0; o >>= 1) mx = fmaxf(mx, __shfl_xor(mx, o, 32));
    float sum = 0.f;
    for (int s = lane; s < SEQ; s += 32) {
      float e = __expf(srow[s] - mx);
      srow[s] = e;
      sum += e;
    }
#pragma unroll
    for (int o = 16; o > 0; o >>= 1) sum += __shfl_xor(sum, o, 32);
    const float inv = 1.f / sum;
    for (int s = lane; s < SEQ; s += 32) srow[s] *= inv;
  }
  __syncthreads();

  // ---- Phase 3: C[h,l] = sum_s P[h,s]*KV[s,l]; 8 waves split latent dim ----
  const int lbase = wave * 64;            // 4 N-tiles of 16 per wave
  v8f cacc[4];
  const v8f vzero = {};
#pragma unroll
  for (int t = 0; t < 4; ++t) cacc[t] = vzero;

  const float* prow = sS + (size_t)row * SEQ + half * 8;  // A: probs (K = s)
  for (int st2 = 0; st2 < SEQ / 32; ++st2) {
    v16bf a = frag_row_f32(prow + st2 * 32);
    if (st2 + 1 < SEQ / 32)  // prefetch next KV K-slab for this wave's columns
      __builtin_prefetch(kvb + (size_t)((st2 + 1) * 32 + half * 8) * LORA + lbase, 0, 1);
#pragma unroll
    for (int t = 0; t < 4; ++t) {
      const int l = lbase + t * 16 + row;
      v16bf bb = frag_strided_f32(kvb + l, LORA, st2 * 32 + half * 8);
      cacc[t] = wmma_bf16(a, bb, cacc[t]);
    }
  }
  const int mb = (lane >> 4) * 8;
#pragma unroll
  for (int t = 0; t < 4; ++t) {
    const int l = lbase + t * 16 + (lane & 15);
#pragma unroll
    for (int e = 0; e < 8; ++e)
      ctx[((size_t)b * NH + h0 + mb + e) * LORA + l] = cacc[t][e];
  }
}

// ---------------------------------------------------------------------------
// Kernel 4: out[b,h,d] = sum_l ctx[b,h,l] * w_vc[h,d,l]  (both K-contiguous)
// grid (2, 128, 2), block 128.
// ---------------------------------------------------------------------------
__global__ void k_out(const float* __restrict__ ctx,
                      const float* __restrict__ wvc,
                      float* __restrict__ out) {
  const int lane = threadIdx.x & 31, wave = threadIdx.x >> 5;
  const int row = lane & 15, half = lane >> 4;
  const int d0 = (blockIdx.x * 4 + wave) * 16;  // 0..127
  const int h = blockIdx.y;
  const int m0 = blockIdx.z * 16;

  const float* arow = ctx + ((size_t)(m0 + row) * NH + h) * LORA + half * 8;
  const float* brow = wvc + ((size_t)h * HD + d0 + row) * LORA + half * 8;

  v8f acc = {};
#pragma unroll 4
  for (int kt = 0; kt < LORA / 32; ++kt) {
    v16bf a = frag_row_f32(arow + kt * 32);
    v16bf b = frag_row_f32(brow + kt * 32);
    acc = wmma_bf16(a, b, acc);
  }
  const int n = d0 + (lane & 15);
  const int mb = m0 + (lane >> 4) * 8;
#pragma unroll
  for (int e = 0; e < 8; ++e)
    out[(size_t)(mb + e) * NHHD + (size_t)h * HD + n] = acc[e];
}

// ---------------------------------------------------------------------------
extern "C" void kernel_launch(void* const* d_in, const int* in_sizes, int n_in,
                              void* d_out, int out_size, void* d_ws, size_t ws_size,
                              hipStream_t stream) {
  (void)in_sizes; (void)n_in; (void)out_size; (void)ws_size;
  const float* hs  = (const float*)d_in[0];  // [32, 1536]
  const float* kv  = (const float*)d_in[1];  // [32, 4096, 512]
  const float* qw  = (const float*)d_in[2];  // [16384, 1536]
  const float* qb  = (const float*)d_in[3];  // [16384]
  const float* wkc = (const float*)d_in[4];  // [128, 128, 512]
  const float* wvc = (const float*)d_in[5];  // [128, 128, 512]
  float* out = (float*)d_out;                // [32, 16384]

  // Workspace layout (13 MB total):
  char* ws = (char*)d_ws;
  bf16*  qbf = (bf16*)ws;                         // 32*16384 bf16   = 1 MB
  bf16*  qa  = (bf16*)(ws + (1u << 20));          // 32*128*512 bf16 = 4 MB
  float* ctx = (float*)(ws + (5u << 20));         // 32*128*512 f32  = 8 MB

  k_qproj <<<dim3(NHHD / 64, 2),      128, 0, stream>>>(hs, qw, qb, qbf);
  k_absorb<<<dim3(LORA / 64, NH, 2),  128, 0, stream>>>(qbf, wkc, qa);
  k_attn  <<<dim3(NH / 16, BSZ),      256, 16 * SEQ * sizeof(float), stream>>>(qa, kv, ctx);
  k_out   <<<dim3(HD / 64, NH, 2),    128, 0, stream>>>(ctx, wvc, out);
}
